// GCN_22969485100000
// MI455X (gfx1250) — matrix operand order repaired and checked by reading, not compiled, not run on hardware
//
#include <hip/hip_runtime.h>
#include <stdint.h>

typedef __attribute__((ext_vector_type(16))) __bf16 v16bf;
typedef __attribute__((ext_vector_type(8)))  float  v8f;

#define IN_F   128
#define HID_F  128
#define OUT_F  64
#define BN_EPS 1e-5f

// ---------- helpers ----------
static __device__ __forceinline__ unsigned short f2bf(float f) {
    union { float f; unsigned int u; } c; c.f = f;
    unsigned int u = c.u;
    // round-to-nearest-even to bf16
    unsigned int r = (u + 0x7FFFu + ((u >> 16) & 1u)) >> 16;
    return (unsigned short)r;
}

static __device__ __forceinline__ void atomAddF32(float* p, float v) {
    // native global_atomic_add_f32 (coarse-grained HBM/L2 memory)
    unsafeAtomicAdd(p, v);
}

// ---------- degree / normalization ----------
__global__ void k_deg_init(float* deg, int n) {
    int i = blockIdx.x * blockDim.x + threadIdx.x;
    if (i < n) deg[i] = 1.0f;                 // self-loop contributes 1
}

__global__ void k_deg_count(const int* __restrict__ dst, float* deg, int e) {
    int i = blockIdx.x * blockDim.x + threadIdx.x;
    if (i < e) atomAddF32(&deg[dst[i]], 1.0f);
}

__global__ void k_deg_finish(float* deg, int n) {
    int i = blockIdx.x * blockDim.x + threadIdx.x;
    if (i < n) deg[i] = rsqrtf(deg[i]);       // deg > 0 always (self loop)
}

// ---------- precision conversion ----------
__global__ void k_f32_to_bf16(const float* __restrict__ x,
                              unsigned short* __restrict__ y, int n) {
    int i = blockIdx.x * blockDim.x + threadIdx.x;
    if (i < n) y[i] = f2bf(x[i]);
}

// Pack W[k][n] (row-major f32, K=128) into WMMA B-fragment order (bf16):
// idx = ((nT*4 + kb)*32 + lane)*16 + e ; n = nT*16+(lane&15) ; k = kb*32+8*(lane>>4)+(e&7)+16*(e>>3)
__global__ void k_pack_w(const float* __restrict__ W,
                         unsigned short* __restrict__ Bp, int Nout) {
    int t = blockIdx.x * blockDim.x + threadIdx.x;
    if (t >= 128 * Nout) return;
    int e    = t & 15;
    int lane = (t >> 4) & 31;
    int kb   = (t >> 9) & 3;       // K = 128 -> 4 k-blocks of 32
    int nT   = t >> 11;
    int ncol = nT * 16 + (lane & 15);
    int krow = kb * 32 + ((lane >> 4) << 3) + (e & 7) + ((e >> 3) << 4);
    Bp[t] = f2bf(W[krow * Nout + ncol]);
}

// ---------- WMMA GEMM ----------
// One wave per 16-row m-tile; computes ALL Nout/16 output tiles so the A
// fragment is loaded once per k-step. K = 128 fixed.
template <int NOUT>
__global__ void k_gemm_bf16(const unsigned short* __restrict__ A,
                            const unsigned short* __restrict__ Bp,
                            float* __restrict__ C, int nRows) {
    constexpr int NT = NOUT / 16;
    int lane  = threadIdx.x & 31;
    int wave  = threadIdx.x >> 5;
    int mTile = blockIdx.x * (blockDim.x >> 5) + wave;
    if (mTile * 16 >= nRows) return;

    int  row  = mTile * 16 + (lane & 15);
    bool full = (mTile * 16 + 16) <= nRows;
    int  rowc = row < nRows ? row : nRows - 1; // clamp for loads on ragged tile
    const unsigned short* arow = A + (size_t)rowc * 128 + ((lane >> 4) << 3);

    union Frag { v16bf v; uint4 u[2]; };
    v8f acc[NT];
#pragma unroll
    for (int nt = 0; nt < NT; ++nt) acc[nt] = (v8f){};

#pragma unroll
    for (int kb = 0; kb < 4; ++kb) {
        Frag a;
        a.u[0] = *(const uint4*)(arow + kb * 32);        // K = kb*32+khi    .. +7
        a.u[1] = *(const uint4*)(arow + kb * 32 + 16);   // K = kb*32+khi+16 .. +23
#pragma unroll
        for (int nt = 0; nt < NT; ++nt) {
            Frag b;
            const unsigned short* bp = Bp + ((size_t)(nt * 4 + kb) * 32 + lane) * 16;
            b.u[0] = *(const uint4*)(bp);
            b.u[1] = *(const uint4*)(bp + 8);
            acc[nt] = __builtin_amdgcn_wmma_f32_16x16x32_bf16(
                          false, a.v, false, b.v, (short)0, acc[nt], false, false);
        }
    }

    int r0   = mTile * 16 + ((lane >> 4) << 3);
    int colb = lane & 15;
    if (full) {
#pragma unroll
        for (int nt = 0; nt < NT; ++nt) {
            float* cp = C + (size_t)r0 * NOUT + nt * 16 + colb;
#pragma unroll
            for (int i = 0; i < 8; ++i) cp[(size_t)i * NOUT] = acc[nt][i];
        }
    } else {
#pragma unroll
        for (int nt = 0; nt < NT; ++nt) {
#pragma unroll
            for (int i = 0; i < 8; ++i) {
                int r = r0 + i;
                if (r < nRows) C[(size_t)r * NOUT + nt * 16 + colb] = acc[nt][i];
            }
        }
    }
}

// ---------- self-loop + bias (full overwrite of h) ----------
template <int C>
__global__ void k_self_bias(const float* __restrict__ xw,
                            const float* __restrict__ dinv,
                            const float* __restrict__ bias,
                            float* __restrict__ h, int n) {
    int t = blockIdx.x * blockDim.x + threadIdx.x;
    if (t >= n * C) return;
    int i = t / C;                 // C is a power of two -> shift
    int c = t & (C - 1);
    float di = dinv[i];
    h[t] = bias[c] + di * di * xw[t];
}

// ---------- edge scatter-add: h[dst] += dinv[s]*dinv[d] * xw[src] ----------
template <int C>
__global__ void k_scatter(const int* __restrict__ src, const int* __restrict__ dst,
                          const float* __restrict__ dinv,
                          const float* __restrict__ xw,
                          float* __restrict__ h, int e) {
    constexpr int Q4 = C >> 2;
    int t = blockIdx.x * blockDim.x + threadIdx.x;
    if (t >= e * Q4) return;
    int ed = t / Q4;               // power of two -> shift
    int q  = t & (Q4 - 1);
    int s  = src[ed];
    int d  = dst[ed];
    float nrm = dinv[s] * dinv[d];
    float4 v = *(const float4*)(xw + (size_t)s * C + q * 4);
    float* hp = h + (size_t)d * C + q * 4;
    atomAddF32(hp + 0, nrm * v.x);
    atomAddF32(hp + 1, nrm * v.y);
    atomAddF32(hp + 2, nrm * v.z);
    atomAddF32(hp + 3, nrm * v.w);
}

// ---------- fused BatchNorm + ReLU -> bf16 features for next layer ----------
template <int C>
__global__ void k_bn_relu_bf16(const float* __restrict__ h,
                               const float* __restrict__ g,
                               const float* __restrict__ beta,
                               const float* __restrict__ m,
                               const float* __restrict__ v,
                               unsigned short* __restrict__ out, int n) {
    int t = blockIdx.x * blockDim.x + threadIdx.x;
    if (t >= n * C) return;
    int c = t & (C - 1);
    float sc = g[c] * rsqrtf(v[c] + BN_EPS);
    float y  = (h[t] - m[c]) * sc + beta[c];
    out[t] = f2bf(fmaxf(y, 0.0f));
}

// ---------- host launch ----------
extern "C" void kernel_launch(void* const* d_in, const int* in_sizes, int n_in,
                              void* d_out, int out_size, void* d_ws, size_t ws_size,
                              hipStream_t stream) {
    (void)n_in; (void)out_size; (void)ws_size;
    const float* x   = (const float*)d_in[0];
    const int*   ei  = (const int*)  d_in[1];
    const float* W0  = (const float*)d_in[2];
    const float* b0  = (const float*)d_in[3];
    const float* g0  = (const float*)d_in[4];
    const float* be0 = (const float*)d_in[5];
    const float* m0  = (const float*)d_in[6];
    const float* v0  = (const float*)d_in[7];
    const float* W1  = (const float*)d_in[8];
    const float* b1  = (const float*)d_in[9];
    const float* g1  = (const float*)d_in[10];
    const float* be1 = (const float*)d_in[11];
    const float* m1  = (const float*)d_in[12];
    const float* v1  = (const float*)d_in[13];
    const float* W2  = (const float*)d_in[14];
    const float* b2  = (const float*)d_in[15];

    const int N = in_sizes[0] / IN_F;
    const int E = in_sizes[1] / 2;
    const int* src = ei;
    const int* dst = ei + E;

    // workspace carve-up (256B aligned)
    char* ws = (char*)d_ws;
    size_t off = 0;
    auto carve = [&](size_t bytes) -> void* {
        void* p = ws + off;
        off += (bytes + 255) & ~(size_t)255;
        return p;
    };
    float*          dinv = (float*)         carve((size_t)N * 4);
    unsigned short* hbf  = (unsigned short*)carve((size_t)N * HID_F * 2);
    unsigned short* Wp   = (unsigned short*)carve((size_t)HID_F * HID_F * 2);
    float*          xw   = (float*)         carve((size_t)N * HID_F * 4);
    float*          h    = (float*)         carve((size_t)N * HID_F * 4);

    const int B = 256;
    auto blocks = [&](long long work) { return (int)((work + B - 1) / B); };

    const int mTiles     = (N + 15) / 16;
    const int gemmBlocks = (mTiles + 7) / 8;   // 8 waves / block

    // normalization coefficients (graph-invariant across layers)
    k_deg_init  <<<blocks(N), B, 0, stream>>>(dinv, N);
    k_deg_count <<<blocks(E), B, 0, stream>>>(dst, dinv, E);
    k_deg_finish<<<blocks(N), B, 0, stream>>>(dinv, N);

    // input features -> bf16
    k_f32_to_bf16<<<blocks((long long)N * IN_F), B, 0, stream>>>(x, hbf, N * IN_F);

    // ---- layer 0: conv(IN->HID) + BN + ReLU ----
    k_pack_w<<<blocks(128 * HID_F), B, 0, stream>>>(W0, Wp, HID_F);
    k_gemm_bf16<HID_F><<<gemmBlocks, 256, 0, stream>>>(hbf, Wp, xw, N);
    k_self_bias<HID_F><<<blocks((long long)N * HID_F), B, 0, stream>>>(xw, dinv, b0, h, N);
    k_scatter<HID_F><<<blocks((long long)E * (HID_F / 4)), B, 0, stream>>>(src, dst, dinv, xw, h, E);
    k_bn_relu_bf16<HID_F><<<blocks((long long)N * HID_F), B, 0, stream>>>(h, g0, be0, m0, v0, hbf, N);

    // ---- layer 1: conv(HID->HID) + BN + ReLU ----
    k_pack_w<<<blocks(128 * HID_F), B, 0, stream>>>(W1, Wp, HID_F);
    k_gemm_bf16<HID_F><<<gemmBlocks, 256, 0, stream>>>(hbf, Wp, xw, N);
    k_self_bias<HID_F><<<blocks((long long)N * HID_F), B, 0, stream>>>(xw, dinv, b1, h, N);
    k_scatter<HID_F><<<blocks((long long)E * (HID_F / 4)), B, 0, stream>>>(src, dst, dinv, xw, h, E);
    k_bn_relu_bf16<HID_F><<<blocks((long long)N * HID_F), B, 0, stream>>>(h, g1, be1, m1, v1, hbf, N);

    // ---- layer 2: conv(HID->OUT) -> d_out (logits) ----
    float* out = (float*)d_out;
    k_pack_w<<<blocks(128 * OUT_F), B, 0, stream>>>(W2, Wp, OUT_F);
    k_gemm_bf16<OUT_F><<<gemmBlocks, 256, 0, stream>>>(hbf, Wp, xw, N);
    k_self_bias<OUT_F><<<blocks((long long)N * OUT_F), B, 0, stream>>>(xw, dinv, b2, out, N);
    k_scatter<OUT_F><<<blocks((long long)E * (OUT_F / 4)), B, 0, stream>>>(src, dst, dinv, xw, out, E);
}
